// MultiheadLatentAttention_26027501814395
// MI455X (gfx1250) — matrix operand order repaired and checked by reading, not compiled, not run on hardware
//
#include <hip/hip_runtime.h>
#include <hip/hip_bf16.h>

// ---------------------------------------------------------------------------
// MLA forward for MI455X (gfx1250, wave32, WMMA bf16 16x16x32, f32 accum)
// GEMMs: async-to-LDS double-buffered tiles; attention: flash w/ WMMA.
// ---------------------------------------------------------------------------

#define Bc 2
#define Sc 2048
#define Ec 2048
#define Hc 16
#define Lc 512
#define Rc 1024
#define Dc 128          // E / H
#define RDc 64          // R / H

typedef __attribute__((ext_vector_type(16))) __bf16 v16bf;
typedef __attribute__((ext_vector_type(8)))  __bf16 v8bf;
typedef __attribute__((ext_vector_type(8)))  float  v8f;
typedef int v4i_vs __attribute__((vector_size(4 * sizeof(int))));

#if __has_builtin(__builtin_amdgcn_global_load_async_to_lds_b128)
#define USE_ASYNC_LDS 1
#else
#define USE_ASYNC_LDS 0
#endif

// 16-byte global -> LDS copy. Async (ASYNCcnt, VGPR-bypassing) when available.
__device__ __forceinline__ void cp16_g2l(const __bf16* g, __bf16* l) {
#if USE_ASYNC_LDS
  __builtin_amdgcn_global_load_async_to_lds_b128(
      (__attribute__((address_space(1))) v4i_vs*)g,
      (__attribute__((address_space(3))) v4i_vs*)l, 0, 0);
#else
  *(v8bf*)l = *(const v8bf*)g;
#endif
}

__device__ __forceinline__ void wait_async_all() {
#if USE_ASYNC_LDS
#if __has_builtin(__builtin_amdgcn_s_wait_asynccnt)
  __builtin_amdgcn_s_wait_asynccnt(0);
#else
  asm volatile("s_wait_asynccnt 0" ::: "memory");
#endif
#endif
}

// ---------------------------------------------------------------------------
// Fragment helpers (layouts per CDNA5 ISA 7.12.2, wave32)
// ---------------------------------------------------------------------------

__device__ __forceinline__ v16bf combine8(v8bf lo, v8bf hi) {
  v16bf r;
#pragma unroll
  for (int i = 0; i < 8; ++i) { r[i] = lo[i]; r[i + 8] = hi[i]; }
  return r;
}

// A-matrix 16x32 bf16 fragment. A points at tile row 0; lda in elements.
// lane 0-15: row = lane,    K = kb+[0..7]  (v0-3), kb+[16..23] (v4-7)
// lane16-31: row = lane-16, K = kb+[8..15] (v0-3), kb+[24..31] (v4-7)
__device__ __forceinline__ v16bf load_a_frag(const __bf16* __restrict__ A,
                                             int lda, int kb, int lane) {
  int row   = lane & 15;
  int kbase = kb + ((lane >> 4) << 3);
  const __bf16* p = A + (size_t)row * lda + kbase;
  v8bf lo = *(const v8bf*)(p);
  v8bf hi = *(const v8bf*)(p + 16);
  return combine8(lo, hi);
}

// B-matrix 32x16 bf16 fragment from row-major W (N x K); per-lane column.
// lane 0-15: col = lane,    K = kb+[0..7]  (v0-3), kb+[16..23] (v4-7)
// lane16-31: col = lane-16, K = kb+[8..15] (v0-3), kb+[24..31] (v4-7)
__device__ __forceinline__ v16bf load_b_frag(const __bf16* __restrict__ W,
                                             int ldk, int kb, int lane) {
  int col   = lane & 15;
  int kbase = kb + ((lane >> 4) << 3);
  const __bf16* p = W + (size_t)col * ldk + kbase;
  v8bf lo = *(const v8bf*)(p);
  v8bf hi = *(const v8bf*)(p + 16);
  return combine8(lo, hi);
}

__device__ __forceinline__ v8f wmma_bf16(v16bf a, v16bf b, v8f c) {
  return __builtin_amdgcn_wmma_f32_16x16x32_bf16(
      /*neg_a=*/false, a, /*neg_b=*/false, b,
      /*c_mod=*/(short)0, c, /*reuse_a=*/false, /*reuse_b=*/false);
}

// 16-lane row reductions (C/D fragment rows live across lanes 0-15 / 16-31)
__device__ __forceinline__ float rowmax16(float v) {
#pragma unroll
  for (int m = 1; m < 16; m <<= 1) v = fmaxf(v, __shfl_xor(v, m, 32));
  return v;
}
__device__ __forceinline__ float rowsum16(float v) {
#pragma unroll
  for (int m = 1; m < 16; m <<= 1) v += __shfl_xor(v, m, 32);
  return v;
}

// ---------------------------------------------------------------------------
// fp32 -> bf16 conversion
// ---------------------------------------------------------------------------
__global__ void f32_to_bf16_kernel(const float* __restrict__ in,
                                   __bf16* __restrict__ out, size_t n) {
  size_t i = (size_t)blockIdx.x * blockDim.x + threadIdx.x;
  size_t stride = (size_t)gridDim.x * blockDim.x;
  for (; i < n; i += stride) out[i] = (__bf16)in[i];
}

// ---------------------------------------------------------------------------
// Generic GEMM: C = A (MxK, row-major bf16) * W^T (W is NxK row-major bf16)
// Block tile 64(M) x 128(N); 8 waves, each a 32x32 micro-tile (4 WMMA / Kstep)
// A/B K-slices (32 wide) staged into LDS with async-to-LDS, double buffered.
// mode 0: bf16 row-major C   mode 1: f32 row-major C
// mode 2: bf16 "V-transposed" store: (b, h, d, s) with m=b*S+s, n=h*D+d
// ---------------------------------------------------------------------------
__global__ __launch_bounds__(256) void gemm_bf16_kernel(
    const __bf16* __restrict__ A, const __bf16* __restrict__ W,
    void* __restrict__ Cout, int M, int N, int K, int mode) {
  __shared__ __bf16 Asm[2][64 * 32];    //  4 KB per buffer
  __shared__ __bf16 Bsm[2][128 * 32];   //  8 KB per buffer

  const int tid  = threadIdx.x;
  const int lane = tid & 31;
  const int wid  = tid >> 5;                   // 0..7
  const int mblk = blockIdx.x * 64;
  const int nblk = blockIdx.y * 128;
  const int wm = (wid >> 2) * 32;              // wave micro-tile origin in LDS
  const int wn = (wid & 3) * 32;

  // Stage one 32-wide K-slice of the block tile into LDS buffer `buf`.
  auto stage = [&](int buf, int kb) {
    const int row = tid >> 2;                  // 0..63
    const int cc  = (tid & 3) * 8;             // 0,8,16,24
    cp16_g2l(A + (size_t)(mblk + row) * K + kb + cc, &Asm[buf][row * 32 + cc]);
    cp16_g2l(W + (size_t)(nblk + row) * K + kb + cc, &Bsm[buf][row * 32 + cc]);
    cp16_g2l(W + (size_t)(nblk + row + 64) * K + kb + cc,
             &Bsm[buf][(row + 64) * 32 + cc]);
  };

  stage(0, 0);
  wait_async_all();
  __syncthreads();

  v8f acc00 = {}, acc01 = {}, acc10 = {}, acc11 = {};
  int buf = 0;
  for (int kb = 0; kb < K; kb += 32) {
    if (kb + 32 < K) stage(buf ^ 1, kb + 32);  // overlap next-slice staging

    v16bf a0 = load_a_frag(&Asm[buf][wm * 32],        32, 0, lane);
    v16bf a1 = load_a_frag(&Asm[buf][(wm + 16) * 32], 32, 0, lane);
    v16bf b0 = load_b_frag(&Bsm[buf][wn * 32],        32, 0, lane);
    v16bf b1 = load_b_frag(&Bsm[buf][(wn + 16) * 32], 32, 0, lane);
    acc00 = wmma_bf16(a0, b0, acc00);
    acc01 = wmma_bf16(a0, b1, acc01);
    acc10 = wmma_bf16(a1, b0, acc10);
    acc11 = wmma_bf16(a1, b1, acc11);

    wait_async_all();                          // next buffer landed
    __syncthreads();                           // everyone done with `buf`
    buf ^= 1;
  }

  const int rbase = ((lane >> 4) << 3);
  const int cbase = (lane & 15);
#pragma unroll
  for (int mi = 0; mi < 2; ++mi) {
#pragma unroll
    for (int ni = 0; ni < 2; ++ni) {
      v8f acc = (mi == 0) ? (ni == 0 ? acc00 : acc01)
                          : (ni == 0 ? acc10 : acc11);
#pragma unroll
      for (int j = 0; j < 8; ++j) {
        int row = mblk + wm + mi * 16 + rbase + j;
        int col = nblk + wn + ni * 16 + cbase;
        float v = acc[j];
        if (mode == 0) {
          ((__bf16*)Cout)[(size_t)row * N + col] = (__bf16)v;
        } else if (mode == 1) {
          ((float*)Cout)[(size_t)row * N + col] = v;
        } else {
          // V^T per head: (b,h,d,s), row = b*S+s, col = h*D+d
          int b = row >> 11, s = row & (Sc - 1);
          int h = col >> 7,  d = col & (Dc - 1);
          ((__bf16*)Cout)[(((size_t)b * Hc + h) * Dc + d) * Sc + s] = (__bf16)v;
        }
      }
    }
  }
}

// ---------------------------------------------------------------------------
// RoPE (in place) on bf16 tensor laid out (B, S, H, RD)
// ---------------------------------------------------------------------------
__global__ void rope_kernel(__bf16* __restrict__ t, size_t n_half) {
  size_t idx = (size_t)blockIdx.x * blockDim.x + threadIdx.x;
  if (idx >= n_half) return;                    // B*S*H*(RD/2)
  int i = idx & 31;                             // rotary pair index 0..31
  int h = (idx >> 5) & (Hc - 1);
  int s = (int)((idx >> 9) & (Sc - 1));
  int b = (int)(idx >> 20);
  float inv_freq = __expf(-__logf(10000.0f) * (float)(2 * i) / (float)RDc);
  float ang = (float)s * inv_freq;
  float c = __cosf(ang), sn = __sinf(ang);
  __bf16* base = t + ((size_t)(b * Sc + s) * Rc + h * RDc);
  float t1 = (float)base[i];
  float t2 = (float)base[i + 32];
  base[i]      = (__bf16)(t1 * c - t2 * sn);
  base[i + 32] = (__bf16)(t1 * sn + t2 * c);
}

// ---------------------------------------------------------------------------
// Flash attention: grid (S/128, H, B), 8 waves/block, wave = 16 query rows.
// K-contraction = 192 (D=128 nope + RD=64 rope), 32 keys per iteration.
// q,k in (B,S,E) per-head slices; qr,kr in (B,S,R); vt in (B,H,D,S).
// out bf16 in (B,S,E).
// ---------------------------------------------------------------------------
__global__ __launch_bounds__(256) void mla_flash_attn_kernel(
    const __bf16* __restrict__ q,  const __bf16* __restrict__ qr,
    const __bf16* __restrict__ k,  const __bf16* __restrict__ kr,
    const __bf16* __restrict__ vt, __bf16* __restrict__ out) {
  __shared__ __bf16 Plds[8 * 16 * 32];          // per-wave 16x32 bf16 tile

  const int lane = threadIdx.x & 31;
  const int wid  = threadIdx.x >> 5;
  const int b = blockIdx.z, h = blockIdx.y;
  const int sq0 = blockIdx.x * 128 + wid * 16;  // wave's first query row

  const float scale = 0.07216878364870322f;     // 1/sqrt(192)

  // Preload the 6 q A-fragments (K = 192)
  const __bf16* qA  = q  + (size_t)(b * Sc + sq0) * Ec + h * Dc;
  const __bf16* qrA = qr + (size_t)(b * Sc + sq0) * Rc + h * RDc;
  v16bf qf[6];
#pragma unroll
  for (int c = 0; c < 4; ++c) qf[c] = load_a_frag(qA, Ec, c * 32, lane);
  qf[4] = load_a_frag(qrA, Rc, 0,  lane);
  qf[5] = load_a_frag(qrA, Rc, 32, lane);

  v8f O[8];
#pragma unroll
  for (int c = 0; c < 8; ++c) O[c] = (v8f){};
  float mrow[8], lrow[8];
#pragma unroll
  for (int j = 0; j < 8; ++j) { mrow[j] = -1e30f; lrow[j] = 0.0f; }

  const int rbase = ((lane >> 4) << 3);
  const int cbase = (lane & 15);
  __bf16* Pw = &Plds[wid * 16 * 32];
  const __bf16* vbase = vt + ((size_t)(b * Hc + h) * Dc) * Sc;

  const int niter = (blockIdx.x + 1) * 4;       // uniform across the block
  for (int it = 0; it < niter; ++it) {
    const int key0 = it * 32;
    // Prefetch next key tile (one cacheline per lane, rows key0+32..+63)
    if (it + 1 < niter) {
      __builtin_prefetch(k  + (size_t)(b * Sc + key0 + 32 + lane) * Ec + h * Dc,  0, 0);
      __builtin_prefetch(kr + (size_t)(b * Sc + key0 + 32 + lane) * Rc + h * RDc, 0, 0);
    }
    // ---- scores: 16 q-rows x 32 keys ----
    const __bf16* kA0 = k  + (size_t)(b * Sc + key0) * Ec + h * Dc;
    const __bf16* kA1 = k  + (size_t)(b * Sc + key0 + 16) * Ec + h * Dc;
    const __bf16* kr0 = kr + (size_t)(b * Sc + key0) * Rc + h * RDc;
    const __bf16* kr1 = kr + (size_t)(b * Sc + key0 + 16) * Rc + h * RDc;
    v8f s0 = {}, s1 = {};
#pragma unroll
    for (int c = 0; c < 4; ++c) {
      s0 = wmma_bf16(qf[c], load_b_frag(kA0, Ec, c * 32, lane), s0);
      s1 = wmma_bf16(qf[c], load_b_frag(kA1, Ec, c * 32, lane), s1);
    }
#pragma unroll
    for (int c = 0; c < 2; ++c) {
      s0 = wmma_bf16(qf[4 + c], load_b_frag(kr0, Rc, c * 32, lane), s0);
      s1 = wmma_bf16(qf[4 + c], load_b_frag(kr1, Rc, c * 32, lane), s1);
    }
    // ---- scale + causal mask + online softmax ----
    const int col0 = key0 + cbase;
    const int col1 = col0 + 16;
    float p0[8], p1[8], alpha[8];
#pragma unroll
    for (int j = 0; j < 8; ++j) {
      const int row = sq0 + rbase + j;
      float v0 = (col0 <= row) ? s0[j] * scale : -1e30f;
      float v1 = (col1 <= row) ? s1[j] * scale : -1e30f;
      float mj = rowmax16(fmaxf(v0, v1));
      float mnew = fmaxf(mrow[j], mj);
      alpha[j] = __expf(mrow[j] - mnew);
      p0[j] = __expf(v0 - mnew);
      p1[j] = __expf(v1 - mnew);
      lrow[j] = lrow[j] * alpha[j] + rowsum16(p0[j] + p1[j]);
      mrow[j] = mnew;
    }
#pragma unroll
    for (int c = 0; c < 8; ++c)
#pragma unroll
      for (int j = 0; j < 8; ++j) O[c][j] *= alpha[j];

    // ---- relayout P (C-layout f32) -> LDS -> A-fragment bf16 ----
#pragma unroll
    for (int j = 0; j < 8; ++j) {
      Pw[(rbase + j) * 32 + cbase]      = (__bf16)p0[j];
      Pw[(rbase + j) * 32 + 16 + cbase] = (__bf16)p1[j];
    }
    __syncthreads();
    v16bf pfrag = load_a_frag(Pw, 32, 0, lane);
    // ---- O += P @ V, 8 chunks of 16 d-columns, V^T rows are d ----
#pragma unroll
    for (int c = 0; c < 8; ++c) {
      v16bf vfrag = load_b_frag(vbase + (size_t)(c * 16) * Sc, Sc, key0, lane);
      O[c] = wmma_bf16(pfrag, vfrag, O[c]);
    }
    __syncthreads();
  }

  // ---- normalize and store (B,S,E) bf16 ----
  float rinv[8];
#pragma unroll
  for (int j = 0; j < 8; ++j) rinv[j] = 1.0f / lrow[j];
#pragma unroll
  for (int c = 0; c < 8; ++c) {
#pragma unroll
    for (int j = 0; j < 8; ++j) {
      int row = sq0 + rbase + j;
      int d = c * 16 + cbase;
      out[(size_t)(b * Sc + row) * Ec + h * Dc + d] = (__bf16)(O[c][j] * rinv[j]);
    }
  }
}

// ---------------------------------------------------------------------------
// Host-side orchestration
// ---------------------------------------------------------------------------
extern "C" void kernel_launch(void* const* d_in, const int* in_sizes, int n_in,
                              void* d_out, int out_size, void* d_ws, size_t ws_size,
                              hipStream_t stream) {
  (void)in_sizes; (void)n_in; (void)out_size; (void)ws_size;
  const float* x    = (const float*)d_in[0];
  const float* Wq   = (const float*)d_in[1];
  const float* Wdkv = (const float*)d_in[2];
  const float* Wuk  = (const float*)d_in[3];
  const float* Wuv  = (const float*)d_in[4];
  const float* Wqr  = (const float*)d_in[5];
  const float* Wkr  = (const float*)d_in[6];
  const float* Wo   = (const float*)d_in[7];

  const size_t Mrows = (size_t)Bc * Sc;         // 4096
  __bf16* p = (__bf16*)d_ws;
  auto carve = [&](size_t n) { __bf16* r = p; p += n; return r; };
  __bf16* xb    = carve(Mrows * Ec);
  __bf16* Wqb   = carve((size_t)Ec * Ec);
  __bf16* Wdkvb = carve((size_t)Lc * Ec);
  __bf16* Wukb  = carve((size_t)Ec * Lc);
  __bf16* Wuvb  = carve((size_t)Ec * Lc);
  __bf16* Wqrb  = carve((size_t)Rc * Ec);
  __bf16* Wkrb  = carve((size_t)Rc * Ec);
  __bf16* Wob   = carve((size_t)Ec * Ec);
  __bf16* qb    = carve(Mrows * Ec);
  __bf16* ckvb  = carve(Mrows * Lc);
  __bf16* kkb   = carve(Mrows * Ec);
  __bf16* vtb   = carve(Mrows * Ec);            // (B,H,D,S)
  __bf16* qrb   = carve(Mrows * Rc);
  __bf16* krb   = carve(Mrows * Rc);
  __bf16* aob   = carve(Mrows * Ec);

  auto cvt = [&](const float* src, __bf16* dst, size_t n) {
    int blocks = (int)((n + 255) / 256);
    if (blocks > 2048) blocks = 2048;
    f32_to_bf16_kernel<<<blocks, 256, 0, stream>>>(src, dst, n);
  };
  cvt(x,    xb,    Mrows * Ec);
  cvt(Wq,   Wqb,   (size_t)Ec * Ec);
  cvt(Wdkv, Wdkvb, (size_t)Lc * Ec);
  cvt(Wuk,  Wukb,  (size_t)Ec * Lc);
  cvt(Wuv,  Wuvb,  (size_t)Ec * Lc);
  cvt(Wqr,  Wqrb,  (size_t)Rc * Ec);
  cvt(Wkr,  Wkrb,  (size_t)Rc * Ec);
  cvt(Wo,   Wob,   (size_t)Ec * Ec);

  auto gemm = [&](const __bf16* A, const __bf16* W, void* C,
                  int M, int N, int K, int mode) {
    dim3 grid(M / 64, N / 128);
    gemm_bf16_kernel<<<grid, 256, 0, stream>>>(A, W, C, M, N, K, mode);
  };

  gemm(xb,   Wqb,   qb,   (int)Mrows, Ec, Ec, 0);   // q
  gemm(xb,   Wdkvb, ckvb, (int)Mrows, Lc, Ec, 0);   // c_kv
  gemm(ckvb, Wukb,  kkb,  (int)Mrows, Ec, Lc, 0);   // k
  gemm(ckvb, Wuvb,  vtb,  (int)Mrows, Ec, Lc, 2);   // v, stored (b,h,d,s)
  gemm(xb,   Wqrb,  qrb,  (int)Mrows, Rc, Ec, 0);   // q_rope pre
  gemm(xb,   Wkrb,  krb,  (int)Mrows, Rc, Ec, 0);   // k_rope pre

  {
    size_t n_half = Mrows * Hc * (RDc / 2);         // 2,097,152
    int blocks = (int)((n_half + 255) / 256);
    rope_kernel<<<blocks, 256, 0, stream>>>(qrb, n_half);
    rope_kernel<<<blocks, 256, 0, stream>>>(krb, n_half);
  }

  {
    dim3 grid(Sc / 128, Hc, Bc);
    mla_flash_attn_kernel<<<grid, 256, 0, stream>>>(qb, qrb, kkb, krb, vtb, aob);
  }

  gemm(aob, Wob, d_out, (int)Mrows, Ec, Ec, 1);     // final projection (f32)
}
